// PromptSequenceVQ_20392504721504
// MI455X (gfx1250) — compile-verified, matrix operand order checked by reading, not run on hardware
//
#include <hip/hip_runtime.h>
#include <hip/hip_bf16.h>

typedef __attribute__((ext_vector_type(16))) __bf16 v16bf;
typedef __attribute__((ext_vector_type(8)))  float  v8f;

#define BN    131072   // 32*4096 rows
#define DIMD  128
#define KCODE 512
#define ROWS_PER_WG 64

// ---------------------------------------------------------------------------
// Kernel 0: zero the atomic accumulators (deterministic across graph replays)
// ---------------------------------------------------------------------------
__global__ void vq_zero(float* __restrict__ encsum, float* __restrict__ dw,
                        float* __restrict__ loss_acc) {
    int i = blockIdx.x * blockDim.x + threadIdx.x;
    if (i == 0) *loss_acc = 0.0f;
    if (i < KCODE) encsum[i] = 0.0f;
    if (i < KCODE * DIMD) dw[i] = 0.0f;
}

// ---------------------------------------------------------------------------
// Kernel 1: split emb into bf16 hi/lo fragments pre-swizzled into the WMMA
// B-matrix (32x16, 16-bit) lane layout, + exact fp32 ||e_k||^2.
// Fragment addressing: frag[((ct*4+kt)*2+h)*32 + lane][e], e in 0..15 (bf16),
// so each lane reads a contiguous 32B v16bf in the hot loop.
// K mapping (ISA 7.12.2, 16-bit 16x32): VGPR j=e/2:
//   K = kt*32 + (j>=4 ? 16:0) + (lane/16)*8 + (j&3)*2 + (e&1)
// ---------------------------------------------------------------------------
__global__ __launch_bounds__(128) void vq_prep(const float* __restrict__ emb,
                                               __bf16* __restrict__ efrag,
                                               float* __restrict__ enorm) {
    __shared__ float red[128];
    const int c = blockIdx.x;       // code 0..511
    const int t = threadIdx.x;      // 0..127

    // exact fp32 ||e_c||^2
    float x = emb[c * DIMD + t];
    red[t] = x * x;
    __syncthreads();
    for (int s = 64; s > 0; s >>= 1) {
        if (t < s) red[t] += red[t + s];
        __syncthreads();
    }
    if (t == 0) enorm[c] = red[0];

    // this block fills all fragment slots where (ct = c/16, lane%16 = c%16)
    const int ct = c >> 4;
    const int kt = t >> 5;          // 0..3
    const int lh = (t >> 4) & 1;    // lane half (0 -> lanes 0-15, 1 -> 16-31)
    const int e  = t & 15;          // element within v16bf
    const int lane = lh * 16 + (c & 15);
    const int j = e >> 1;
    const int Kk = kt * 32 + ((j >= 4) ? 16 : 0) + lh * 8 + (j & 3) * 2 + (e & 1);

    float xv = emb[c * DIMD + Kk];
    __bf16 hi = (__bf16)xv;
    __bf16 lo = (__bf16)(xv - (float)hi);
    int base = (((ct * 4 + kt) * 2 + 0) * 32 + lane) * 16 + e;
    efrag[base]       = hi;   // h = 0
    efrag[base + 512] = lo;   // h = 1 (one 32x16 fragment = 512 bf16)
}

// ---------------------------------------------------------------------------
// Kernel 2: main VQ kernel. 128 threads = 4 waves; WG covers 64 rows of z.
// Split-bf16 WMMA: S = zhi*ehi + zhi*elo + zlo*ehi  (fp32 accumulate)
//   - three independent accumulator chains (hazard/ILP cover)
//   - double-buffered B-fragment prefetch (overlap L2 fetch with WMMAs)
// dist(m,k) = ||e_k||^2 - 2*S[m][k]   (||z||^2 constant per row for argmin)
// ---------------------------------------------------------------------------
__global__ __launch_bounds__(128) void vq_main(
    const float*  __restrict__ z,      const float* __restrict__ emb,
    const __bf16* __restrict__ efrag,  const float* __restrict__ enorm,
    float* __restrict__ q_out,  float* __restrict__ idx_out,
    float* __restrict__ encsum, float* __restrict__ dw,
    float* __restrict__ loss_acc)
{
    __shared__ alignas(32) __bf16 zfrag[4 * 4 * 2 * 32 * 16];  // 32 KB
    __shared__ float red[128];
    __shared__ float znorm[ROWS_PER_WG];

    const int tid = threadIdx.x;
    const long rowbase_wg = (long)blockIdx.x * ROWS_PER_WG;

    // ---- phase 1: convert 64 z rows -> bf16 hi/lo in A-fragment LDS layout
    {
        const int r = tid >> 1;          // local row 0..63
        const int hc = tid & 1;          // which 64-col half
        const int rb = r >> 4, mm = r & 15;
        const float* src = z + (rowbase_wg + r) * DIMD + hc * 64;
        float ss = 0.0f;
#pragma unroll
        for (int i = 0; i < 64; i += 4) {
            float4 v = *(const float4*)(src + i);
            float xs[4] = {v.x, v.y, v.z, v.w};
#pragma unroll
            for (int u = 0; u < 4; ++u) {
                int k = hc * 64 + i + u;
                float xv = xs[u];
                ss += xv * xv;
                __bf16 hi = (__bf16)xv;
                __bf16 lo = (__bf16)(xv - (float)hi);
                int kt = k >> 5, kk = k & 31;
                int j = ((kk >= 16) ? 4 : 0) + ((kk & 7) >> 1);
                int e = (j << 1) | (kk & 1);
                int lane = (((kk >> 3) & 1) << 4) | mm;
                int base = (((rb * 4 + kt) * 2 + 0) * 32 + lane) * 16 + e;
                zfrag[base]       = hi;
                zfrag[base + 512] = lo;
            }
        }
        red[tid] = ss;
    }
    __syncthreads();
    if (tid < ROWS_PER_WG) znorm[tid] = red[2 * tid] + red[2 * tid + 1];
    __syncthreads();

    // ---- phase 2: per-wave 16-row block, WMMA over 32 code tiles
    const int wave = tid >> 5;
    const int lane = tid & 31;

    v16bf ah[4], al[4];
#pragma unroll
    for (int kt = 0; kt < 4; ++kt) {
        int off = (((wave * 4 + kt) * 2 + 0) * 32 + lane) * 16;
        ah[kt] = *(const v16bf*)&zfrag[off];
        al[kt] = *(const v16bf*)&zfrag[off + 512];
    }

    float minval[8];
    int   minidx[8];
#pragma unroll
    for (int v = 0; v < 8; ++v) { minval[v] = 3.4e38f; minidx[v] = 0; }

    const int col = lane & 15;

    auto loadB = [&](int ct, v16bf* bh, v16bf* bl) {
#pragma unroll
        for (int kt = 0; kt < 4; ++kt) {
            const __bf16* bp = efrag + ((((ct * 4 + kt) * 2 + 0) * 32 + lane) * 16);
            bh[kt] = *(const v16bf*)bp;
            bl[kt] = *(const v16bf*)(bp + 512);
        }
    };

    auto processTile = [&](int ct, const v16bf* bh, const v16bf* bl) {
        // three independent WMMA chains: zhi*ehi, zhi*elo, zlo*ehi
        v8f c0 = {}, c1 = {}, c2 = {};
#pragma unroll
        for (int kt = 0; kt < 4; ++kt) {
            c0 = __builtin_amdgcn_wmma_f32_16x16x32_bf16(false, ah[kt], false, bh[kt],
                                                         (short)0, c0, false, false);
            c1 = __builtin_amdgcn_wmma_f32_16x16x32_bf16(false, ah[kt], false, bl[kt],
                                                         (short)0, c1, false, false);
            c2 = __builtin_amdgcn_wmma_f32_16x16x32_bf16(false, al[kt], false, bh[kt],
                                                         (short)0, c2, false, false);
        }
        int code = ct * 16 + col;
        float en = enorm[code];
#pragma unroll
        for (int v = 0; v < 8; ++v) {
            float d = en - 2.0f * c0[v];
            d = __builtin_fmaf(-2.0f, c1[v], d);
            d = __builtin_fmaf(-2.0f, c2[v], d);
            if (d < minval[v]) { minval[v] = d; minidx[v] = code; }
        }
    };

    // double-buffered loop over the 32 code tiles (two tiles per trip)
    v16bf bhA[4], blA[4], bhB[4], blB[4];
    loadB(0, bhA, blA);
#pragma unroll 1
    for (int ct = 0; ct < 32; ct += 2) {
        loadB(ct + 1, bhB, blB);              // prefetch odd tile
        processTile(ct, bhA, blA);
        loadB((ct + 2) & 31, bhA, blA);       // prefetch next even tile (clamped)
        processTile(ct + 1, bhB, blB);
    }

    // ---- argmin butterfly across the 16-lane groups (ties -> lowest index)
#pragma unroll
    for (int m = 1; m < 16; m <<= 1) {
#pragma unroll
        for (int v = 0; v < 8; ++v) {
            float ov = __shfl_xor(minval[v], m, 16);
            int   oi = __shfl_xor(minidx[v], m, 16);
            if (ov < minval[v] || (ov == minval[v] && oi < minidx[v])) {
                minval[v] = ov; minidx[v] = oi;
            }
        }
    }

    const long rowbase_w = rowbase_wg + wave * 16;

    // ---- per-row scalars: indices, code counts, loss partials
    if (lane == 0 || lane == 16) {
        const int mbase = (lane >> 4) * 8;   // lane0 -> rows 0-7, lane16 -> 8-15
        float lsum = 0.0f;
#pragma unroll
        for (int v = 0; v < 8; ++v) {
            long grow = rowbase_w + mbase + v;
            idx_out[grow] = (float)minidx[v];
            atomicAdd(&encsum[minidx[v]], 1.0f);
            // ||e_idx - z||^2 = ||z||^2 + (||e||^2 - 2 z.e) = znorm + minval
            lsum += znorm[wave * 16 + mbase + v] + minval[v];
        }
        atomicAdd(loss_acc, lsum);
    }

    // ---- broadcast indices, gather quantized rows (exact fp32), scatter dw
    int idxs[16];
#pragma unroll
    for (int v = 0; v < 8; ++v) {
        idxs[v]     = __shfl(minidx[v], 0, 32);
        idxs[8 + v] = __shfl(minidx[v], 16, 32);
    }
#pragma unroll 4
    for (int m = 0; m < 16; ++m) {
        int c = idxs[m];
        long grow = rowbase_w + m;
        float4 e4 = *(const float4*)(emb + c * DIMD + lane * 4);
        *(float4*)(q_out + grow * DIMD + lane * 4) = e4;   // quantized_st == emb[idx]
        float4 z4 = *(const float4*)(z + grow * DIMD + lane * 4);
        float* dwp = dw + c * DIMD + lane * 4;
        atomicAdd(dwp + 0, z4.x);
        atomicAdd(dwp + 1, z4.y);
        atomicAdd(dwp + 2, z4.z);
        atomicAdd(dwp + 3, z4.w);
    }
}

// ---------------------------------------------------------------------------
// Kernel 3: scalars — vq_loss and perplexity
// ---------------------------------------------------------------------------
__global__ __launch_bounds__(256) void vq_final(const float* __restrict__ encsum,
                                                const float* __restrict__ loss_acc,
                                                float* __restrict__ vq_loss,
                                                float* __restrict__ perp) {
    __shared__ float red[256];
    const int t = threadIdx.x;
    float s = 0.0f;
    for (int k = t; k < KCODE; k += 256) {
        float p = encsum[k] * (1.0f / (float)BN);
        s += p * logf(p + 1e-10f);
    }
    red[t] = s;
    __syncthreads();
    for (int st = 128; st > 0; st >>= 1) {
        if (t < st) red[t] += red[t + st];
        __syncthreads();
    }
    if (t == 0) {
        *perp = expf(-red[0]);
        // q_latent + 0.25*e_latent, numerically (1.25)*mean((e[idx]-z)^2)
        *vq_loss = 1.25f * (*loss_acc) * (1.0f / ((float)BN * (float)DIMD));
    }
}

// ---------------------------------------------------------------------------
extern "C" void kernel_launch(void* const* d_in, const int* in_sizes, int n_in,
                              void* d_out, int out_size, void* d_ws, size_t ws_size,
                              hipStream_t stream) {
    const float* z   = (const float*)d_in[0];   // (32,4096,128) f32
    const float* emb = (const float*)d_in[1];   // (512,128) f32

    float* out      = (float*)d_out;
    float* q_out    = out;                               // 16777216
    float* idx_out  = out + (long)BN * DIMD;             // 131072
    float* loss_out = idx_out + BN;                      // 1
    float* perp_out = loss_out + 1;                      // 1
    float* encsum   = perp_out + 1;                      // 512
    float* dw       = encsum + KCODE;                    // 65536

    char* ws = (char*)d_ws;
    float*  loss_acc = (float*)ws;                       // 1 f32 (padded)
    __bf16* efrag    = (__bf16*)(ws + 256);              // 131072 bf16 = 256 KB
    float*  enorm    = (float*)(ws + 256 + 262144);      // 512 f32

    vq_zero <<<(KCODE * DIMD + 255) / 256, 256, 0, stream>>>(encsum, dw, loss_acc);
    vq_prep <<<KCODE, 128, 0, stream>>>(emb, efrag, enorm);
    vq_main <<<BN / ROWS_PER_WG, 128, 0, stream>>>(z, emb, efrag, enorm,
                                                   q_out, idx_out, encsum, dw, loss_acc);
    vq_final<<<1, 256, 0, stream>>>(encsum, loss_acc, loss_out, perp_out);
}
